// RecurrentMoE_9964324127040
// MI455X (gfx1250) — matrix operand order repaired and checked by reading, not compiled, not run on hardware
//
#include <hip/hip_runtime.h>

typedef __attribute__((ext_vector_type(16))) __bf16 v16bf;
typedef __attribute__((ext_vector_type(8)))  float  v8f;

#define B_    8
#define T_    2048
#define D_    1024
#define H_    16
#define S_    8
#define E_    8
#define KSEL_ 2
#define DF_   4096
#define HD_   64
#define L_    (S_ + T_)     /* 2056 */
#define R_    (B_ * L_)     /* 16448 */
#define EPS_  1e-5f
#define KC_   256           /* K-chunk per block in skinny WMMA GEMM */

// Async global->LDS copy path (gfx1250): guarded so absence falls back to the
// proven synchronous staging. Evaluated per compilation pass (host pass -> 0).
// Probe-derived signature: (int4 AS1* src, int4 AS3* dst, imm offset, imm cpol)
#if defined(__gfx1250__) && __has_builtin(__builtin_amdgcn_global_load_async_to_lds_b128) && __has_builtin(__builtin_amdgcn_s_wait_asynccnt)
#define ASYNC_LDS 1
typedef __attribute__((ext_vector_type(4))) int ai4_t;
typedef __attribute__((address_space(1))) ai4_t as1_i4;
typedef __attribute__((address_space(3))) ai4_t as3_i4;
#else
#define ASYNC_LDS 0
#endif

__device__ __forceinline__ float gelu_exact(float x) {
    return 0.5f * x * (1.0f + erff(x * 0.70710678118654752f));
}

// ---------------------------------------------------------------------------
// LN(concat(state, x)) -> bf16, one block per row of [B, L, D]
// ---------------------------------------------------------------------------
__global__ __launch_bounds__(256) void ln_cast_kv_kernel(
    const float* __restrict__ x, const float* __restrict__ state,
    const float* __restrict__ g, const float* __restrict__ beta,
    __bf16* __restrict__ out)
{
    __shared__ float red[256];
    const int row = blockIdx.x;
    const int b = row / L_, i = row % L_;
    const float* src = (i < S_) ? state + ((size_t)b * S_ + i) * D_
                                : x + ((size_t)b * T_ + (i - S_)) * D_;
    const int t = threadIdx.x;
    float v[4]; float s = 0.f;
    #pragma unroll
    for (int j = 0; j < 4; ++j) { v[j] = src[t + 256 * j]; s += v[j]; }
    red[t] = s; __syncthreads();
    for (int st = 128; st > 0; st >>= 1) { if (t < st) red[t] += red[t + st]; __syncthreads(); }
    const float mean = red[0] * (1.f / D_); __syncthreads();
    float s2 = 0.f;
    #pragma unroll
    for (int j = 0; j < 4; ++j) { float d = v[j] - mean; s2 += d * d; }
    red[t] = s2; __syncthreads();
    for (int st = 128; st > 0; st >>= 1) { if (t < st) red[t] += red[t + st]; __syncthreads(); }
    const float inv = rsqrtf(red[0] * (1.f / D_) + EPS_);
    __bf16* orow = out + (size_t)row * D_;
    #pragma unroll
    for (int j = 0; j < 4; ++j) {
        int d = t + 256 * j;
        orow[d] = (__bf16)((v[j] - mean) * inv * g[d] + beta[d]);
    }
}

// ---------------------------------------------------------------------------
// LN over 64 contiguous rows of D_ (fp32 out)
// ---------------------------------------------------------------------------
__global__ __launch_bounds__(256) void ln_rows_kernel(
    const float* __restrict__ src, const float* __restrict__ g,
    const float* __restrict__ beta, float* __restrict__ out)
{
    __shared__ float red[256];
    const int row = blockIdx.x;
    const int t = threadIdx.x;
    const float* p = src + (size_t)row * D_;
    float v[4]; float s = 0.f;
    #pragma unroll
    for (int j = 0; j < 4; ++j) { v[j] = p[t + 256 * j]; s += v[j]; }
    red[t] = s; __syncthreads();
    for (int st = 128; st > 0; st >>= 1) { if (t < st) red[t] += red[t + st]; __syncthreads(); }
    const float mean = red[0] * (1.f / D_); __syncthreads();
    float s2 = 0.f;
    #pragma unroll
    for (int j = 0; j < 4; ++j) { float d = v[j] - mean; s2 += d * d; }
    red[t] = s2; __syncthreads();
    for (int st = 128; st > 0; st >>= 1) { if (t < st) red[t] += red[t + st]; __syncthreads(); }
    const float inv = rsqrtf(red[0] * (1.f / D_) + EPS_);
    float* orow = out + (size_t)row * D_;
    #pragma unroll
    for (int j = 0; j < 4; ++j) {
        int d = t + 256 * j;
        orow[d] = (v[j] - mean) * inv * g[d] + beta[d];
    }
}

// ---------------------------------------------------------------------------
// fp32 -> bf16 elementwise cast
// ---------------------------------------------------------------------------
__global__ __launch_bounds__(256) void cast_bf16_kernel(
    const float* __restrict__ in, __bf16* __restrict__ out, int n)
{
    for (int i = blockIdx.x * 256 + threadIdx.x; i < n; i += gridDim.x * 256)
        out[i] = (__bf16)in[i];
}

// ---------------------------------------------------------------------------
// zero fill
// ---------------------------------------------------------------------------
__global__ __launch_bounds__(256) void fill_zero_kernel(float* __restrict__ p, long long n)
{
    for (long long i = (long long)blockIdx.x * 256 + threadIdx.x; i < n;
         i += (long long)gridDim.x * 256)
        p[i] = 0.f;
}

// ---------------------------------------------------------------------------
// bf16 WMMA GEMM: C[Rr,Nn] = A[Rr,Kd] @ W[Kd,Nn] + bias   (bf16 in, f32 acc)
// 128x64 tile per 256-thread (8-wave) block; each wave 16x64 via 4 WMMAs
// A tile staged via async global->LDS copy when available.
// ---------------------------------------------------------------------------
__global__ __launch_bounds__(256) void gemm_bf16_kernel(
    const __bf16* __restrict__ A, const __bf16* __restrict__ W,
    const float* __restrict__ bias, __bf16* __restrict__ C,
    int Rr, int Kd, int Nn)
{
    __shared__ __bf16 As[128][40];   // padded stride: 80B/row, 16B-aligned
    __shared__ __bf16 Bt[64][40];    // transposed: Bt[n][k]
    const int tid  = threadIdx.x;
    const int wave = tid >> 5;
    const int lane = tid & 31;
    const int m0 = blockIdx.x * 128;
    const int n0 = blockIdx.y * 64;

    v8f acc[4] = {};

    for (int k0 = 0; k0 < Kd; k0 += 32) {
        // A tile 128x32 bf16 = 512 x b128 -> 2 per thread
        #pragma unroll
        for (int it = 0; it < 2; ++it) {
            int i = tid + it * 256;
            int r = i >> 2;
            int c = (i & 3) << 3;
            int gr = m0 + r; if (gr >= Rr) gr = Rr - 1;
            const __bf16* gp = A + (size_t)gr * Kd + k0 + c;
#if ASYNC_LDS
            __builtin_amdgcn_global_load_async_to_lds_b128(
                (as1_i4*)const_cast<__bf16*>(gp),
                (as3_i4*)&As[r][c], 0, 0);
#else
            *reinterpret_cast<uint4*>(&As[r][c]) =
                *reinterpret_cast<const uint4*>(gp);
#endif
        }
        // B tile 32x64 -> transposed into Bt[64][.]
        {
            int r = tid >> 3;            // k row 0..31
            int c = (tid & 7) << 3;      // n col start
            uint4 v = *reinterpret_cast<const uint4*>(W + (size_t)(k0 + r) * Nn + n0 + c);
            const __bf16* p = reinterpret_cast<const __bf16*>(&v);
            #pragma unroll
            for (int j = 0; j < 8; ++j) Bt[c + j][r] = p[j];
        }
        if (k0 + 32 < Kd) {  // global_prefetch_b8 for next K tile
            __builtin_prefetch(A + (size_t)(m0 + (tid >> 1)) * Kd + k0 + 32, 0, 0);
            __builtin_prefetch(W + (size_t)(k0 + 32 + (tid >> 3)) * Nn + n0, 0, 0);
        }
#if ASYNC_LDS
        __builtin_amdgcn_s_wait_asynccnt(0);
#endif
        __syncthreads();

        // A fragment: lane holds row (wave*16 + lane&15); contiguous K (16h + s)
        union { v16bf v; unsigned u[8]; } af;
        const int am = (wave << 4) + (lane & 15);
        const int kb = (lane >> 4) << 4;           // 0 or 16
        #pragma unroll
        for (int vv = 0; vv < 8; ++vv) {
            int kk = kb + (vv << 1);
            af.u[vv] = *reinterpret_cast<const unsigned*>(&As[am][kk]);
        }
        const int kbb = (lane >> 4) << 4;          // 0 or 16
        #pragma unroll
        for (int nt = 0; nt < 4; ++nt) {
            union { v16bf v; unsigned u[8]; } bfr;
            const int bn = (nt << 4) + (lane & 15);
            #pragma unroll
            for (int vv = 0; vv < 8; ++vv)
                bfr.u[vv] = *reinterpret_cast<const unsigned*>(&Bt[bn][kbb + (vv << 1)]);
            acc[nt] = __builtin_amdgcn_wmma_f32_16x16x32_bf16(
                false, af.v, false, bfr.v, (short)0, acc[nt], false, false);
        }
        __syncthreads();
    }

    const int rbase = m0 + (wave << 4) + ((lane >> 4) << 3);
    #pragma unroll
    for (int nt = 0; nt < 4; ++nt) {
        const int col = n0 + (nt << 4) + (lane & 15);
        const float bv = bias[col];
        #pragma unroll
        for (int r = 0; r < 8; ++r) {
            int row = rbase + r;
            if (row < Rr) C[(size_t)row * Nn + col] = (__bf16)(acc[nt][r] + bv);
        }
    }
}

// ---------------------------------------------------------------------------
// Skinny WMMA GEMM (M=64) with K-split and atomic f32 accumulation.
// Grid: (Nn/64, Kd/KC_, E).  A_e = A + e*Astride, W_e = W + e*Wstride,
// C_e = Cacc + e*Cstride.  fp32 inputs converted to bf16 while staging.
// If wescale != null, K-partials are scaled by wescale[row/S_ * E_ + e]
// (linear, so valid on partials).  Cacc must be pre-initialized.
// ---------------------------------------------------------------------------
__global__ __launch_bounds__(256) void wmma_skinny_kernel(
    const float* __restrict__ A, size_t Astride,
    const float* __restrict__ W, size_t Wstride,
    float* __restrict__ Cacc, size_t Cstride,
    const float* __restrict__ wescale,
    int Kd, int Nn)
{
    __shared__ __bf16 As[64][40];
    __shared__ __bf16 Bt[64][40];
    const int tid = threadIdx.x, wave = tid >> 5, lane = tid & 31;
    const int n0 = blockIdx.x * 64;
    const int k0base = blockIdx.y * KC_;
    const int e = blockIdx.z;
    const float* Ae = A + (size_t)e * Astride;
    const float* We = W + (size_t)e * Wstride;
    float* Ce = Cacc + (size_t)e * Cstride;

    const int mt = wave >> 1;     // 0..3 : M-tile of 16 rows
    const int nh = wave & 1;      // 0..1 : 32-col half

    v8f acc[2] = {};

    for (int k0 = k0base; k0 < k0base + KC_; k0 += 32) {
        // stage A chunk [64][32] fp32 -> bf16 (2048 elems, 8/thread)
        #pragma unroll
        for (int it = 0; it < 8; ++it) {
            int i = tid + it * 256;
            int r = i >> 5, c = i & 31;
            As[r][c] = (__bf16)Ae[(size_t)r * Kd + k0 + c];
        }
        // stage W chunk [32][64] fp32 -> Bt[64][.] bf16 transposed
        #pragma unroll
        for (int it = 0; it < 8; ++it) {
            int i = tid + it * 256;
            int r = i >> 6, c = i & 63;
            Bt[c][r] = (__bf16)We[(size_t)(k0 + r) * Nn + n0 + c];
        }
        __syncthreads();
        union { v16bf v; unsigned u[8]; } af;
        const int am = (mt << 4) + (lane & 15);
        const int kb = (lane >> 4) << 4;
        #pragma unroll
        for (int vv = 0; vv < 8; ++vv) {
            int kk = kb + (vv << 1);
            af.u[vv] = *reinterpret_cast<const unsigned*>(&As[am][kk]);
        }
        const int kbb = (lane >> 4) << 4;
        #pragma unroll
        for (int nt = 0; nt < 2; ++nt) {
            union { v16bf v; unsigned u[8]; } bfr;
            const int bn = (nh << 5) + (nt << 4) + (lane & 15);
            #pragma unroll
            for (int vv = 0; vv < 8; ++vv)
                bfr.u[vv] = *reinterpret_cast<const unsigned*>(&Bt[bn][kbb + (vv << 1)]);
            acc[nt] = __builtin_amdgcn_wmma_f32_16x16x32_bf16(
                false, af.v, false, bfr.v, (short)0, acc[nt], false, false);
        }
        __syncthreads();
    }
    const int rbase = (mt << 4) + ((lane >> 4) << 3);
    #pragma unroll
    for (int nt = 0; nt < 2; ++nt) {
        const int col = n0 + (nh << 5) + (nt << 4) + (lane & 15);
        #pragma unroll
        for (int r = 0; r < 8; ++r) {
            const int row = rbase + r;
            float v = acc[nt][r];
            if (wescale) v *= wescale[(row >> 3) * E_ + e];
            atomicAdd(&Ce[(size_t)row * Nn + col], v);
        }
    }
}

// ---------------------------------------------------------------------------
// Epilogue: out[i] = maybe_gelu(acc[i] + bias[grp*bias_estride + col]) (+res)
// rows may be 64 (grp=0) or E*64 (grp = expert index)
// ---------------------------------------------------------------------------
__global__ __launch_bounds__(256) void bias_act_kernel(
    const float* __restrict__ acc, const float* __restrict__ bias, int bias_estride,
    const float* __restrict__ res, float* __restrict__ out,
    int rows, int Nn, int do_gelu)
{
    const long long total = (long long)rows * Nn;
    for (long long i = (long long)blockIdx.x * 256 + threadIdx.x; i < total;
         i += (long long)gridDim.x * 256) {
        int col = (int)(i % Nn);
        int grp = (int)(i / ((long long)64 * Nn));
        float v = acc[i] + bias[(size_t)grp * bias_estride + col];
        if (do_gelu) v = gelu_exact(v);
        if (res) v += res[i];
        out[i] = v;
    }
}

// ---------------------------------------------------------------------------
// latacc[m][n] = sum_e we[m/S_][e] * exp_b2[e][n]   (folds MoE bias term)
// ---------------------------------------------------------------------------
__global__ __launch_bounds__(256) void init_moe_bias_kernel(
    const float* __restrict__ we, const float* __restrict__ b2,
    float* __restrict__ latacc)
{
    int i = blockIdx.x * 256 + threadIdx.x;
    if (i >= 64 * D_) return;
    int m = i >> 10, n = i & (D_ - 1);
    float s = 0.f;
    #pragma unroll
    for (int e = 0; e < E_; ++e) s += we[(m >> 3) * E_ + e] * b2[(size_t)e * D_ + n];
    latacc[i] = s;
}

// ---------------------------------------------------------------------------
// Attention scores: scores[b,h,s,k] = (qh[b,s,h,:] . kh[b,k,h,:]) / 8
// ---------------------------------------------------------------------------
__global__ __launch_bounds__(256) void attn_scores_kernel(
    const float* __restrict__ qh, const __bf16* __restrict__ kh,
    float* __restrict__ scores)
{
    __shared__ float qS[S_][HD_];
    const int bh = blockIdx.x, b = bh / H_, h = bh % H_;
    for (int j = threadIdx.x; j < S_ * HD_; j += 256) {
        int s = j >> 6, hd = j & 63;
        qS[s][hd] = qh[((size_t)(b * S_ + s)) * D_ + h * HD_ + hd];
    }
    __syncthreads();
    const int k = blockIdx.y * 256 + threadIdx.x;
    if (k >= L_) return;
    const __bf16* krow = kh + ((size_t)(b * L_ + k)) * D_ + h * HD_;
    float sc[S_];
    #pragma unroll
    for (int s = 0; s < S_; ++s) sc[s] = 0.f;
    #pragma unroll
    for (int c = 0; c < 8; ++c) {
        uint4 v = *reinterpret_cast<const uint4*>(krow + c * 8);
        const __bf16* p = reinterpret_cast<const __bf16*>(&v);
        #pragma unroll
        for (int j = 0; j < 8; ++j) {
            float kv = (float)p[j];
            #pragma unroll
            for (int s = 0; s < S_; ++s) sc[s] = fmaf(qS[s][c * 8 + j], kv, sc[s]);
        }
    }
    #pragma unroll
    for (int s = 0; s < S_; ++s)
        scores[((size_t)bh * S_ + s) * L_ + k] = 0.125f * sc[s];
}

// ---------------------------------------------------------------------------
// Row softmax over L_ (one block per (b,h,s) row)
// ---------------------------------------------------------------------------
__global__ __launch_bounds__(256) void softmax_kernel(float* __restrict__ scores)
{
    __shared__ float red[256];
    const int t = threadIdx.x;
    float* row = scores + (size_t)blockIdx.x * L_;
    float v[9]; float mx = -3.4e38f;
    #pragma unroll
    for (int j = 0; j < 9; ++j) {
        int k = t + 256 * j;
        v[j] = (k < L_) ? row[k] : -3.4e38f;
        mx = fmaxf(mx, v[j]);
    }
    red[t] = mx; __syncthreads();
    for (int st = 128; st > 0; st >>= 1) { if (t < st) red[t] = fmaxf(red[t], red[t + st]); __syncthreads(); }
    mx = red[0]; __syncthreads();
    float sm = 0.f;
    #pragma unroll
    for (int j = 0; j < 9; ++j) { v[j] = __expf(v[j] - mx); sm += v[j]; }
    red[t] = sm; __syncthreads();
    for (int st = 128; st > 0; st >>= 1) { if (t < st) red[t] += red[t + st]; __syncthreads(); }
    const float inv = 1.f / red[0];
    #pragma unroll
    for (int j = 0; j < 9; ++j) {
        int k = t + 256 * j;
        if (k < L_) row[k] = v[j] * inv;
    }
}

// ---------------------------------------------------------------------------
// AV: attn[b,s,h*64+hd] = sum_k att[b,h,s,k] * vh[b,k,h,hd]
// V tile staged via async global->LDS copy when available.
// ---------------------------------------------------------------------------
__global__ __launch_bounds__(256) void attn_av_kernel(
    const float* __restrict__ scores, const __bf16* __restrict__ vh,
    float* __restrict__ attn)
{
    __shared__ float  attS[S_][128];   // 4 KB
    __shared__ __bf16 vhS[128][HD_];   // 16 KB
    const int bh = blockIdx.x, b = bh / H_, h = bh % H_, t = threadIdx.x;
    const int s0 = t >> 6, hd0 = t & 63;
    const int o2 = t + 256;
    const int s1 = o2 >> 6, hd1 = o2 & 63;
    float a0 = 0.f, a1 = 0.f;
    for (int kt = 0; kt < L_; kt += 128) {
        for (int j = t; j < S_ * 128; j += 256) {
            int s = j >> 7, kk = j & 127;
            int k = kt + kk;
            attS[s][kk] = (k < L_) ? scores[((size_t)bh * S_ + s) * L_ + k] : 0.f;
        }
        for (int j = t; j < 128 * 8; j += 256) {
            int r = j >> 3, c = (j & 7) << 3;
            int k = kt + r; if (k >= L_) k = L_ - 1;
            const __bf16* gp = vh + ((size_t)(b * L_ + k)) * D_ + h * HD_ + c;
#if ASYNC_LDS
            __builtin_amdgcn_global_load_async_to_lds_b128(
                (as1_i4*)const_cast<__bf16*>(gp),
                (as3_i4*)&vhS[r][c], 0, 0);
#else
            *reinterpret_cast<uint4*>(&vhS[r][c]) =
                *reinterpret_cast<const uint4*>(gp);
#endif
        }
#if ASYNC_LDS
        __builtin_amdgcn_s_wait_asynccnt(0);
#endif
        __syncthreads();
        #pragma unroll 4
        for (int kk = 0; kk < 128; ++kk) {
            a0 = fmaf(attS[s0][kk], (float)vhS[kk][hd0], a0);
            a1 = fmaf(attS[s1][kk], (float)vhS[kk][hd1], a1);
        }
        __syncthreads();
    }
    attn[((size_t)(b * S_ + s0)) * D_ + h * HD_ + hd0] = a0;
    attn[((size_t)(b * S_ + s1)) * D_ + h * HD_ + hd1] = a1;
}

// ---------------------------------------------------------------------------
// Gating: pooled = LN(mean_s latent); top-2 of pooled@wg; we[b,e]; gidx out
// ---------------------------------------------------------------------------
__global__ __launch_bounds__(256) void gate_kernel(
    const float* __restrict__ latent, const float* __restrict__ g,
    const float* __restrict__ beta, const float* __restrict__ wg,
    float* __restrict__ we, int* __restrict__ gidx_out)
{
    __shared__ float pool[D_];
    __shared__ float red[256];
    __shared__ float logits[E_];
    const int b = blockIdx.x, t = threadIdx.x;
    #pragma unroll
    for (int j = 0; j < 4; ++j) {
        int d = t + 256 * j;
        float s = 0.f;
        for (int ss = 0; ss < S_; ++ss) s += latent[((size_t)(b * S_ + ss)) * D_ + d];
        pool[d] = s * (1.f / S_);
    }
    __syncthreads();
    float s = 0.f;
    #pragma unroll
    for (int j = 0; j < 4; ++j) s += pool[t + 256 * j];
    red[t] = s; __syncthreads();
    for (int st = 128; st > 0; st >>= 1) { if (t < st) red[t] += red[t + st]; __syncthreads(); }
    const float mean = red[0] * (1.f / D_); __syncthreads();
    float s2 = 0.f;
    #pragma unroll
    for (int j = 0; j < 4; ++j) { float d = pool[t + 256 * j] - mean; s2 += d * d; }
    red[t] = s2; __syncthreads();
    for (int st = 128; st > 0; st >>= 1) { if (t < st) red[t] += red[t + st]; __syncthreads(); }
    const float inv = rsqrtf(red[0] * (1.f / D_) + EPS_); __syncthreads();
    float pe[E_];
    #pragma unroll
    for (int e = 0; e < E_; ++e) pe[e] = 0.f;
    #pragma unroll
    for (int j = 0; j < 4; ++j) {
        int d = t + 256 * j;
        float pl = (pool[d] - mean) * inv * g[d] + beta[d];
        #pragma unroll
        for (int e = 0; e < E_; ++e) pe[e] = fmaf(pl, wg[(size_t)d * E_ + e], pe[e]);
    }
    for (int e = 0; e < E_; ++e) {
        red[t] = pe[e]; __syncthreads();
        for (int st = 128; st > 0; st >>= 1) { if (t < st) red[t] += red[t + st]; __syncthreads(); }
        if (t == 0) logits[e] = red[0];
        __syncthreads();
    }
    if (t == 0) {
        int i0 = 0; float v0 = logits[0];
        for (int e = 1; e < E_; ++e) if (logits[e] > v0) { v0 = logits[e]; i0 = e; }
        int i1 = -1; float v1 = -3.4e38f;
        for (int e = 0; e < E_; ++e) if (e != i0 && logits[e] > v1) { v1 = logits[e]; i1 = e; }
        float ex = __expf(v1 - v0);
        float g0 = 1.f / (1.f + ex), g1 = ex / (1.f + ex);
        for (int e = 0; e < E_; ++e) we[b * E_ + e] = 0.f;
        we[b * E_ + i0] = g0;
        we[b * E_ + i1] = g1;
        gidx_out[b * KSEL_ + 0] = i0;
        gidx_out[b * KSEL_ + 1] = i1;
    }
}

// ---------------------------------------------------------------------------
// Slot write-back: slog = LN(latent)@wslot; top-2; state_out, latent, y, sidx
// ---------------------------------------------------------------------------
__global__ __launch_bounds__(256) void writeback_kernel(
    const float* __restrict__ latent, const float* __restrict__ state,
    const float* __restrict__ g, const float* __restrict__ beta,
    const float* __restrict__ wslot,
    float* __restrict__ y_out, float* __restrict__ lat_out,
    float* __restrict__ st_out, int* __restrict__ sidx_out)
{
    __shared__ float red[256];
    __shared__ float slog[S_];
    __shared__ float wsarr[S_];
    const int b = blockIdx.x, t = threadIdx.x;
    for (int sidx = 0; sidx < S_; ++sidx) {
        const float* rowp = latent + ((size_t)(b * S_ + sidx)) * D_;
        float v[4]; float sm = 0.f;
        #pragma unroll
        for (int j = 0; j < 4; ++j) { v[j] = rowp[t + 256 * j]; sm += v[j]; }
        red[t] = sm; __syncthreads();
        for (int st = 128; st > 0; st >>= 1) { if (t < st) red[t] += red[t + st]; __syncthreads(); }
        const float mean = red[0] * (1.f / D_); __syncthreads();
        float s2 = 0.f;
        #pragma unroll
        for (int j = 0; j < 4; ++j) { float d = v[j] - mean; s2 += d * d; }
        red[t] = s2; __syncthreads();
        for (int st = 128; st > 0; st >>= 1) { if (t < st) red[t] += red[t + st]; __syncthreads(); }
        const float inv = rsqrtf(red[0] * (1.f / D_) + EPS_); __syncthreads();
        float dp = 0.f;
        #pragma unroll
        for (int j = 0; j < 4; ++j) {
            int d = t + 256 * j;
            dp = fmaf((v[j] - mean) * inv * g[d] + beta[d], wslot[d], dp);
        }
        red[t] = dp; __syncthreads();
        for (int st = 128; st > 0; st >>= 1) { if (t < st) red[t] += red[t + st]; __syncthreads(); }
        if (t == 0) slog[sidx] = red[0];
        __syncthreads();
    }
    if (t == 0) {
        int i0 = 0; float v0 = slog[0];
        for (int s = 1; s < S_; ++s) if (slog[s] > v0) { v0 = slog[s]; i0 = s; }
        int i1 = -1; float v1 = -3.4e38f;
        for (int s = 0; s < S_; ++s) if (s != i0 && slog[s] > v1) { v1 = slog[s]; i1 = s; }
        for (int s = 0; s < S_; ++s) wsarr[s] = 0.f;
        wsarr[i0] += v0;
        wsarr[i1] += v1;
        sidx_out[b * 2 + 0] = i0;
        sidx_out[b * 2 + 1] = i1;
    }
    __syncthreads();
    #pragma unroll
    for (int j = 0; j < 4; ++j) {
        int d = t + 256 * j;
        float ys = 0.f;
        for (int s = 0; s < S_; ++s) {
            size_t idx = ((size_t)(b * S_ + s)) * D_ + d;
            float lv = latent[idx];
            ys += lv;
            st_out[idx]  = state[idx] + wsarr[s] * lv;
            lat_out[idx] = lv;
        }
        y_out[(size_t)b * D_ + d] = ys * (1.f / S_);
    }
}

// ---------------------------------------------------------------------------
extern "C" void kernel_launch(void* const* d_in, const int* in_sizes, int n_in,
                              void* d_out, int out_size, void* d_ws, size_t ws_size,
                              hipStream_t stream) {
    const float* x        = (const float*)d_in[0];
    const float* state    = (const float*)d_in[1];
    const float* lnq_g    = (const float*)d_in[2];
    const float* lnq_b    = (const float*)d_in[3];
    const float* lnkv_g   = (const float*)d_in[4];
    const float* lnkv_b   = (const float*)d_in[5];
    const float* wq       = (const float*)d_in[6];
    const float* bq       = (const float*)d_in[7];
    const float* wk       = (const float*)d_in[8];
    const float* bk       = (const float*)d_in[9];
    const float* wv       = (const float*)d_in[10];
    const float* bv       = (const float*)d_in[11];
    const float* wo       = (const float*)d_in[12];
    const float* bo       = (const float*)d_in[13];
    const float* lnffn_g  = (const float*)d_in[14];
    const float* lnffn_b  = (const float*)d_in[15];
    const float* ffn_w1   = (const float*)d_in[16];
    const float* ffn_b1   = (const float*)d_in[17];
    const float* ffn_w2   = (const float*)d_in[18];
    const float* ffn_b2   = (const float*)d_in[19];
    const float* lnmoe_g  = (const float*)d_in[20];
    const float* lnmoe_b  = (const float*)d_in[21];
    const float* wg       = (const float*)d_in[22];
    const float* exp_w1   = (const float*)d_in[23];
    const float* exp_b1   = (const float*)d_in[24];
    const float* exp_w2   = (const float*)d_in[25];
    const float* exp_b2   = (const float*)d_in[26];
    const float* lnslot_g = (const float*)d_in[27];
    const float* lnslot_b = (const float*)d_in[28];
    const float* wslot    = (const float*)d_in[29];

    // workspace layout
    char* wsb = (char*)d_ws;
    size_t off = 0;
    auto take = [&](size_t bytes) -> char* {
        char* p = wsb + off;
        off += (bytes + 255) & ~(size_t)255;
        return p;
    };
    __bf16* kvln  = (__bf16*)take((size_t)R_ * D_ * 2);
    __bf16* wkb   = (__bf16*)take((size_t)D_ * D_ * 2);
    __bf16* wvb   = (__bf16*)take((size_t)D_ * D_ * 2);
    __bf16* kh    = (__bf16*)take((size_t)R_ * D_ * 2);
    __bf16* vh    = (__bf16*)take((size_t)R_ * D_ * 2);
    float* qln    = (float*)take((size_t)64 * D_ * 4);
    float* qh     = (float*)take((size_t)64 * D_ * 4);
    float* qhacc  = (float*)take((size_t)64 * D_ * 4);
    float* acc1   = (float*)take((size_t)64 * D_ * 4);
    float* scores = (float*)take((size_t)B_ * H_ * S_ * L_ * 4);
    float* attn   = (float*)take((size_t)64 * D_ * 4);
    float* latent1= (float*)take((size_t)64 * D_ * 4);
    float* hf     = (float*)take((size_t)64 * D_ * 4);
    float* ffnacc = (float*)take((size_t)64 * DF_ * 4);
    float* ffnh   = (float*)take((size_t)64 * DF_ * 4);
    float* latent2= (float*)take((size_t)64 * D_ * 4);
    float* webuf  = (float*)take((size_t)B_ * E_ * 4);
    float* h1acc  = (float*)take((size_t)E_ * 64 * DF_ * 4);   // 8 MB
    float* latacc = (float*)take((size_t)64 * D_ * 4);

    // output layout: y[8192] | latent[65536] | state_out[65536] | gidx[16] | sidx[16]
    float* outf     = (float*)d_out;
    float* y_out    = outf;
    float* lat_out  = outf + 8192;
    float* st_out   = outf + 73728;
    int*   gidx_out = (int*)(outf + 139264);
    int*   sidx_out = (int*)(outf + 139280);

    // 1) LN + cast of concat(latent, x)  -> kvln (bf16)
    ln_cast_kv_kernel<<<R_, 256, 0, stream>>>(x, state, lnkv_g, lnkv_b, kvln);

    // 2) weight casts for the big GEMMs
    cast_bf16_kernel<<<1024, 256, 0, stream>>>(wk, wkb, D_ * D_);
    cast_bf16_kernel<<<1024, 256, 0, stream>>>(wv, wvb, D_ * D_);

    // 3) heavy WMMA GEMMs: kh = kvln@wk + bk ; vh = kvln@wv + bv
    dim3 gg((R_ + 127) / 128, D_ / 64);
    gemm_bf16_kernel<<<gg, 256, 0, stream>>>(kvln, wkb, bk, kh, R_, D_, D_);
    gemm_bf16_kernel<<<gg, 256, 0, stream>>>(kvln, wvb, bv, vh, R_, D_, D_);

    // 4) q path: qln = LN(state); qh = qln@wq + bq   (WMMA skinny, K-split)
    ln_rows_kernel<<<64, 256, 0, stream>>>(state, lnq_g, lnq_b, qln);
    fill_zero_kernel<<<256, 256, 0, stream>>>(qhacc, 64 * D_);
    wmma_skinny_kernel<<<dim3(D_ / 64, D_ / KC_, 1), 256, 0, stream>>>(
        qln, 0, wq, 0, qhacc, 0, nullptr, D_, D_);
    bias_act_kernel<<<256, 256, 0, stream>>>(qhacc, bq, 0, nullptr, qh, 64, D_, 0);

    // 5) attention
    dim3 gs(B_ * H_, (L_ + 255) / 256);
    attn_scores_kernel<<<gs, 256, 0, stream>>>(qh, kh, scores);
    softmax_kernel<<<B_ * H_ * S_, 256, 0, stream>>>(scores);
    attn_av_kernel<<<B_ * H_, 256, 0, stream>>>(scores, vh, attn);

    // 6) latent1 = state + attn@wo + bo
    fill_zero_kernel<<<256, 256, 0, stream>>>(acc1, 64 * D_);
    wmma_skinny_kernel<<<dim3(D_ / 64, D_ / KC_, 1), 256, 0, stream>>>(
        attn, 0, wo, 0, acc1, 0, nullptr, D_, D_);
    bias_act_kernel<<<256, 256, 0, stream>>>(acc1, bo, 0, state, latent1, 64, D_, 0);

    // 7) FFN: latent2 = latent1 + gelu(LN(latent1)@w1+b1)@w2+b2
    ln_rows_kernel<<<64, 256, 0, stream>>>(latent1, lnffn_g, lnffn_b, hf);
    fill_zero_kernel<<<1024, 256, 0, stream>>>(ffnacc, 64 * DF_);
    wmma_skinny_kernel<<<dim3(DF_ / 64, D_ / KC_, 1), 256, 0, stream>>>(
        hf, 0, ffn_w1, 0, ffnacc, 0, nullptr, D_, DF_);
    bias_act_kernel<<<1024, 256, 0, stream>>>(ffnacc, ffn_b1, 0, nullptr, ffnh, 64, DF_, 1);
    fill_zero_kernel<<<256, 256, 0, stream>>>(acc1, 64 * D_);
    wmma_skinny_kernel<<<dim3(D_ / 64, DF_ / KC_, 1), 256, 0, stream>>>(
        ffnh, 0, ffn_w2, 0, acc1, 0, nullptr, DF_, D_);
    bias_act_kernel<<<256, 256, 0, stream>>>(acc1, ffn_b2, 0, latent1, latent2, 64, D_, 0);

    // 8) gating (top-2, softmax weights, gidx)
    gate_kernel<<<B_, 256, 0, stream>>>(latent2, lnmoe_g, lnmoe_b, wg, webuf, gidx_out);

    // 9) experts, all 8 in one grid per GEMM stage:
    //    h1 = gelu(latent2 @ w1_e + b1_e)  -> h1acc (E x 64 x DF)
    fill_zero_kernel<<<4096, 256, 0, stream>>>(h1acc, (long long)E_ * 64 * DF_);
    wmma_skinny_kernel<<<dim3(DF_ / 64, D_ / KC_, E_), 256, 0, stream>>>(
        latent2, 0, exp_w1, (size_t)D_ * DF_, h1acc, (size_t)64 * DF_, nullptr, D_, DF_);
    bias_act_kernel<<<4096, 256, 0, stream>>>(h1acc, exp_b1, DF_, nullptr, h1acc,
                                              E_ * 64, DF_, 1);
    //    latent3 = sum_e we[b,e] * (h1_e @ w2_e) + sum_e we[b,e]*b2_e
    init_moe_bias_kernel<<<(64 * D_) / 256, 256, 0, stream>>>(webuf, exp_b2, latacc);
    wmma_skinny_kernel<<<dim3(D_ / 64, DF_ / KC_, E_), 256, 0, stream>>>(
        h1acc, (size_t)64 * DF_, exp_w2, (size_t)DF_ * D_, latacc, 0, webuf, DF_, D_);

    // 10) slot write-back + outputs
    writeback_kernel<<<B_, 256, 0, stream>>>(latacc, state, lnslot_g, lnslot_b, wslot,
                                             y_out, lat_out, st_out, sidx_out);
}
